// W8A8DynamicLinearStatic_64699387347292
// MI455X (gfx1250) — 8-bit matrix operand order repaired and checked by reading, compiled, not run on hardware
//
#include <hip/hip_runtime.h>
#include <hip/hip_bf16.h>
#include <stdint.h>


#define TOKENS 8192
#define D_IN   4096
#define D_OUT  4096

typedef __attribute__((ext_vector_type(2))) int v2i;
typedef __attribute__((ext_vector_type(4))) int v4i;
typedef __attribute__((ext_vector_type(8))) int v8i;

__global__ __launch_bounds__(256)
void quant_kernel(const float* __restrict__ x, int8_t* __restrict__ xq,
                  float* __restrict__ act_scale, float* __restrict__ row_sum) {
    __shared__ float smax[256];
    __shared__ int   ssum[256];

    const int row = blockIdx.x;
    const int tid = threadIdx.x;
    const float* xr = x + (size_t)row * D_IN;

    float4 v[4];
    float mx = 0.0f;
#pragma unroll
    for (int i = 0; i < 4; ++i) {
        v[i] = ((const float4*)xr)[tid * 4 + i];
        mx = fmaxf(mx, fmaxf(fmaxf(fabsf(v[i].x), fabsf(v[i].y)),
                             fmaxf(fabsf(v[i].z), fabsf(v[i].w))));
    }

    smax[tid] = mx;
    __syncthreads();
#pragma unroll
    for (int s = 128; s > 0; s >>= 1) {
        if (tid < s) smax[tid] = fmaxf(smax[tid], smax[tid + s]);
        __syncthreads();
    }
    const float amax  = fmaxf(smax[0], 1e-30f);
    const float scale = smax[0] / 127.0f;
    const float inv   = 127.0f / amax;

    int isum = 0;
    unsigned int pk[4];
#pragma unroll
    for (int i = 0; i < 4; ++i) {
        int q0 = (int)fminf(fmaxf(rintf(v[i].x * inv), -127.0f), 127.0f);
        int q1 = (int)fminf(fmaxf(rintf(v[i].y * inv), -127.0f), 127.0f);
        int q2 = (int)fminf(fmaxf(rintf(v[i].z * inv), -127.0f), 127.0f);
        int q3 = (int)fminf(fmaxf(rintf(v[i].w * inv), -127.0f), 127.0f);
        isum += q0 + q1 + q2 + q3;
        pk[i] = (unsigned int)(q0 & 255) | ((unsigned int)(q1 & 255) << 8) |
                ((unsigned int)(q2 & 255) << 16) | ((unsigned int)(q3 & 255) << 24);
    }
    ((uint4*)(xq + (size_t)row * D_IN))[tid] = make_uint4(pk[0], pk[1], pk[2], pk[3]);

    ssum[tid] = isum;
    __syncthreads();
#pragma unroll
    for (int s = 128; s > 0; s >>= 1) {
        if (tid < s) ssum[tid] += ssum[tid + s];
        __syncthreads();
    }
    if (tid == 0) {
        act_scale[row] = scale;
        row_sum[row]   = (float)ssum[0];
    }
}

__global__ __launch_bounds__(256)
void packw_kernel(const int* __restrict__ w, int8_t* __restrict__ w8) {
    const size_t idx = (size_t)blockIdx.x * 256 + threadIdx.x;
    int4 t = ((const int4*)w)[idx];
    unsigned int p = (unsigned int)(t.x & 255) | ((unsigned int)(t.y & 255) << 8) |
                     ((unsigned int)(t.z & 255) << 16) | ((unsigned int)(t.w & 255) << 24);
    ((unsigned int*)w8)[idx] = p;
}

__global__ __launch_bounds__(256)
void gemm_kernel(const int8_t* __restrict__ xq, const int8_t* __restrict__ w8,
                 const float* __restrict__ act_scale, const float* __restrict__ row_sum,
                 const float* __restrict__ wscale, const float* __restrict__ woff,
                 const float* __restrict__ bias, float* __restrict__ out) {
    const int lane  = threadIdx.x & 31;
    const int wave  = threadIdx.x >> 5;
    const int lmod  = lane & 15;
    const int lhalf = lane >> 4;

    const int nBlk  = blockIdx.x * 128;
    const int mBlk  = blockIdx.y * 128;
    const int mWave = mBlk + (wave >> 1) * 32;
    const int nWave = nBlk + (wave & 1) * 64;

    v8i acc[2][4];
#pragma unroll
    for (int mi = 0; mi < 2; ++mi)
#pragma unroll
        for (int ni = 0; ni < 4; ++ni)
            acc[mi][ni] = (v8i){0, 0, 0, 0, 0, 0, 0, 0};

    const int8_t* aBase = xq + (size_t)(mWave + lmod) * D_IN + lhalf * 16;
    const int8_t* bBase = w8 + (size_t)(nWave + lmod) * D_IN + lhalf * 16;

    for (int k = 0; k < D_IN; k += 64) {
        v8i a[2];
        v8i b[4];
#pragma unroll
        for (int mi = 0; mi < 2; ++mi) {
            const int8_t* p = aBase + (size_t)(mi * 16) * D_IN + k;
#pragma unroll
            for (int j = 0; j < 4; ++j) {
                v2i t = *(const v2i*)(p + (j >> 1) * 32 + (j & 1) * 8);
                a[mi][2 * j]     = t.x;
                a[mi][2 * j + 1] = t.y;
            }
        }
#pragma unroll
        for (int ni = 0; ni < 4; ++ni) {
            const int8_t* p = bBase + (size_t)(ni * 16) * D_IN + k;
            v4i lo = *(const v4i*)(p);
            v4i hi = *(const v4i*)(p + 32);
            b[ni][0] = lo.x; b[ni][1] = lo.y; b[ni][2] = lo.z; b[ni][3] = lo.w;
            b[ni][4] = hi.x; b[ni][5] = hi.y; b[ni][6] = hi.z; b[ni][7] = hi.w;
        }
#pragma unroll
        for (int mi = 0; mi < 2; ++mi)
#pragma unroll
            for (int ni = 0; ni < 4; ++ni)
                acc[mi][ni] = __builtin_amdgcn_wmma_i32_16x16x64_iu8(
                     true, a[mi],  true, b[ni],
                    acc[mi][ni],  false,  false);
    }

#pragma unroll
    for (int ni = 0; ni < 4; ++ni) {
        const int ncol = nWave + ni * 16 + lmod;
        const float ws_ = wscale[ncol];
        const float wo  = woff[ncol];
        const float bs  = bias[ncol];
#pragma unroll
        for (int mi = 0; mi < 2; ++mi) {
#pragma unroll
            for (int r = 0; r < 8; ++r) {
                const int m  = mWave + mi * 16 + lhalf * 8 + r;
                const float asc = act_scale[m];
                const float rs  = row_sum[m];
                const float yv  = ((float)acc[mi][ni][r] + rs * wo) * (asc * ws_) + bs;
                out[(size_t)m * D_OUT + ncol] = yv;
            }
        }
    }
}

extern "C" void kernel_launch(void* const* d_in, const int* in_sizes, int n_in,
                              void* d_out, int out_size, void* d_ws, size_t ws_size,
                              hipStream_t stream) {
    const float* x      = (const float*)d_in[0];
    const int*   weight = (const int*)  d_in[1];
    const float* wscale = (const float*)d_in[2];
    const float* woff   = (const float*)d_in[3];
    const float* bias   = (const float*)d_in[4];
    float*       out    = (float*)d_out;

    char* ws = (char*)d_ws;
    int8_t* xq        = (int8_t*)ws;
    int8_t* w8        = (int8_t*)(ws + (size_t)32 * 1024 * 1024);
    float*  act_scale = (float*) (ws + (size_t)48 * 1024 * 1024);
    float*  row_sum   = (float*) (ws + (size_t)48 * 1024 * 1024 + TOKENS * sizeof(float));

    quant_kernel<<<TOKENS, 256, 0, stream>>>(x, xq, act_scale, row_sum);

    packw_kernel<<<(D_OUT * (size_t)D_IN) / (4 * 256), 256, 0, stream>>>(weight, w8);

    dim3 grid(D_OUT / 128, TOKENS / 128);
    gemm_kernel<<<grid, 256, 0, stream>>>(xq, w8, act_scale, row_sum,
                                          wscale, woff, bias, out);
}
